// AlexNet_7816840479233
// MI455X (gfx1250) — matrix operand order repaired and checked by reading, not compiled, not run on hardware
//
#include <hip/hip_runtime.h>
#include <hip/hip_bf16.h>

// ---------------------------------------------------------------------------
// NormDist-AlexNet with interval bound propagation, MI455X (gfx1250).
//  * distance convs: VALU kernels, patch triple staged in LDS, wave32 per
//    output channel, two-pass stabilized L8 norm, shfl reductions,
//    global_prefetch_b8 on the weight stream (weights are L2-resident).
//  * bound_linear:    v_wmma_f32_16x16x32_f16, M=16 == batch, 3 accumulators
//    per wave (c@W^T, mid@W^T, rad@|W|^T share the B fragment load).
//    Fragment loads are branch-free float4 (global_load_b128) per the CDNA5
//    wave32 VGPR layouts (ISA 7.12.2).
// ---------------------------------------------------------------------------

typedef _Float16 v16h __attribute__((ext_vector_type(16)));
typedef float    v8f  __attribute__((ext_vector_type(8)));

#define MAXF 3456               // largest C*k*k (layers 4/5: 384*3*3)

__device__ __forceinline__ float wave_max32(float v) {
#pragma unroll
  for (int m = 16; m >= 1; m >>= 1) v = fmaxf(v, __shfl_xor(v, m, 32));
  return v;
}
__device__ __forceinline__ float wave_sum32(float v) {
#pragma unroll
  for (int m = 16; m >= 1; m >>= 1) v += __shfl_xor(v, m, 32);
  return v;
}
__device__ __forceinline__ float pow8f(float x) {  // x^8 = ((x^2)^2)^2
  float t = x * x; t = t * t; return t * t;
}
__device__ __forceinline__ float4 ld4(const float* p) {
  return *(const float4*)p;                        // -> global_load_b128
}

// ---------------------------------------------------------------------------
// Lp(p=8) distance conv with IBP bounds.
// grid.x = B*Ho*Wo (one output pixel per block), grid.y = O/8,
// block = 256 (8 waves), one wave per output channel.
// Outputs are norms (>=0) so the reference's following ReLU is a no-op.
// ---------------------------------------------------------------------------
__global__ void normdist_conv_kernel(
    const float* __restrict__ cin, const float* __restrict__ lin,
    const float* __restrict__ hin, const float* __restrict__ wgt,
    float* __restrict__ cout, float* __restrict__ lout, float* __restrict__ hout,
    int B, int C, int H, int W, int O, int k, int s, int pad, int Ho, int Wo)
{
  __shared__ float smem[3 * MAXF];
  const int F = C * k * k;
  float* sc = smem;
  float* sl = smem + F;
  float* sh = smem + 2 * F;

  const int pix = blockIdx.x;
  const int wo = pix % Wo;
  const int t0 = pix / Wo;
  const int ho = t0 % Ho;
  const int b  = t0 / Ho;

  // cooperative patch-triple load into LDS (torch-unfold channel-major order)
  const int kk = k * k;
  for (int f = threadIdx.x; f < F; f += blockDim.x) {
    const int cch = f / kk;
    const int r   = f - cch * kk;
    const int ky  = r / k;
    const int kx  = r - ky * k;
    const int iy  = ho * s - pad + ky;
    const int ix  = wo * s - pad + kx;
    float vc = 0.f, vl = 0.f, vh = 0.f;
    if (iy >= 0 && iy < H && ix >= 0 && ix < W) {
      const size_t idx = (((size_t)b * C + cch) * H + iy) * W + ix;
      vc = cin[idx]; vl = lin[idx]; vh = hin[idx];
    }
    sc[f] = vc; sl[f] = vl; sh[f] = vh;
  }
  __syncthreads();

  const int wid  = threadIdx.x >> 5;
  const int lane = threadIdx.x & 31;
  const int o    = blockIdx.y * (blockDim.x >> 5) + wid;
  if (o >= O) return;
  const float* __restrict__ wrow = wgt + (size_t)o * F;

  // pass 1: per-lane maxes of the three distance variants
  float mc = 0.f, ml = 0.f, mh = 0.f;
  for (int f = lane; f < F; f += 32) {
    __builtin_prefetch(wrow + f + 256, 0, 1);   // -> global_prefetch_b8
    const float wv = wrow[f];
    const float pc = sc[f], pl = sl[f], ph = sh[f];
    mc = fmaxf(mc, fabsf(pc - wv));
    ml = fmaxf(ml, fmaxf(fmaxf(pl - wv, wv - ph), 0.f));
    mh = fmaxf(mh, fmaxf(fabsf(pl - wv), fabsf(ph - wv)));
  }
  mc = wave_max32(mc); ml = wave_max32(ml); mh = wave_max32(mh);
  const float ic = mc > 0.f ? 1.f / mc : 0.f;
  const float il = ml > 0.f ? 1.f / ml : 0.f;
  const float ih = mh > 0.f ? 1.f / mh : 0.f;

  // pass 2: stabilized sum of (d/m)^8  (weights re-read hit L2)
  float Sc = 0.f, Sl = 0.f, Sh = 0.f;
  for (int f = lane; f < F; f += 32) {
    const float wv = wrow[f];
    const float pc = sc[f], pl = sl[f], ph = sh[f];
    Sc += pow8f(fabsf(pc - wv) * ic);
    Sl += pow8f(fmaxf(fmaxf(pl - wv, wv - ph), 0.f) * il);
    Sh += pow8f(fmaxf(fabsf(pl - wv), fabsf(ph - wv)) * ih);
  }
  Sc = wave_sum32(Sc); Sl = wave_sum32(Sl); Sh = wave_sum32(Sh);

  if (lane == 0) {
    const float oc = mc > 0.f ? mc * powf(Sc, 0.125f) : 0.f;
    const float ol = ml > 0.f ? ml * powf(Sl, 0.125f) : 0.f;
    const float oh = mh > 0.f ? mh * powf(Sh, 0.125f) : 0.f;
    const size_t oidx = (((size_t)b * O + o) * Ho + ho) * Wo + wo;
    cout[oidx] = oc; lout[oidx] = ol; hout[oidx] = oh;
  }
}

// ---------------------------------------------------------------------------
// 3x3 stride-2 VALID maxpool on the (c, l, h) triple.
// ---------------------------------------------------------------------------
__global__ void maxpool3x3s2_kernel(
    const float* __restrict__ ci, const float* __restrict__ li,
    const float* __restrict__ hi_,
    float* __restrict__ co, float* __restrict__ lo, float* __restrict__ ho,
    int BC, int H, int W, int Hp, int Wp)
{
  const int idx = blockIdx.x * blockDim.x + threadIdx.x;
  const int total = BC * Hp * Wp;
  if (idx >= total) return;
  const int wp = idx % Wp;
  const int t  = idx / Wp;
  const int hp = t % Hp;
  const int bc = t / Hp;
  float mc = -INFINITY, ml = -INFINITY, mh = -INFINITY;
#pragma unroll
  for (int dy = 0; dy < 3; ++dy)
#pragma unroll
    for (int dx = 0; dx < 3; ++dx) {
      const size_t s = ((size_t)bc * H + (hp * 2 + dy)) * W + (wp * 2 + dx);
      mc = fmaxf(mc, ci[s]); ml = fmaxf(ml, li[s]); mh = fmaxf(mh, hi_[s]);
    }
  const size_t d = ((size_t)bc * Hp + hp) * Wp + wp;
  co[d] = mc; lo[d] = ml; ho[d] = mh;
}

// ---------------------------------------------------------------------------
// bound_linear via WMMA.  A = 16 x K (batch rows), W = [N, K] row major.
//   acc0 = c   @ W^T        acc1 = (l+h)/2 @ W^T       acc2 = (h-l)/2 @ |W|^T
// One wave per 16x16 output tile; K stepped by 32.
// A 16x32 f16 fragment (lane = row M for lane&15): two contiguous 8-float
// runs at K = kb+8*hsel and K = kb+16+8*hsel  -> 4x float4 per tensor.
// B 32x16 f16 fragment (lane = col N for lane&15): two contiguous 8-float
// runs at K = kb+8*hsel and K = kb+16+8*hsel -> 4x float4.  Out-of-range columns (final layer,
// N=10) are handled branch-free with a clamped row pointer + 0/1 lane mask.
// EXEC is all-ones around every WMMA.  final_neg writes (-c, -u, -l).
// ---------------------------------------------------------------------------
__global__ void bound_linear_wmma_kernel(
    const float* __restrict__ cin, const float* __restrict__ lin,
    const float* __restrict__ hin, const float* __restrict__ Wt,
    const float* __restrict__ bias,
    float* __restrict__ cout, float* __restrict__ lout, float* __restrict__ hout,
    float* __restrict__ dout,
    int K, int N, int do_relu, int final_neg)
{
  const int wid  = threadIdx.x >> 5;
  const int lane = threadIdx.x & 31;
  const int tile = blockIdx.x * (blockDim.x >> 5) + wid;
  if (tile * 16 >= N) return;                 // uniform per wave

  const int row  = lane & 15;                 // A row (M) and B column (N)
  const int hsel = lane >> 4;                 // lane-group K-half select

  const int   n       = tile * 16 + row;      // B column owned by this lane
  const float wscale  = (n < N) ? 1.f : 0.f;  // branch-free OOB column mask
  const float* __restrict__ wrow  = Wt  + (size_t)(n < N ? n : 0) * K;
  const float* __restrict__ arow_c = cin + (size_t)row * K;
  const float* __restrict__ arow_l = lin + (size_t)row * K;
  const float* __restrict__ arow_h = hin + (size_t)row * K;

  v8f acc0 = {}, acc1 = {}, acc2 = {};

  for (int kb = 0; kb < K; kb += 32) {
    // ---- A fragments: 8+8 contiguous floats per tensor -> b128 loads ----
    const int g0 = kb + hsel * 8;        // elements 0..7  -> K = g0 + e
    const int g1 = g0 + 16;              // elements 8..15 -> K = g1 + (e-8)
    float4 c04 = ld4(arow_c + g0), c14 = ld4(arow_c + g0 + 4);
    float4 c24 = ld4(arow_c + g1), c34 = ld4(arow_c + g1 + 4);
    float4 l04 = ld4(arow_l + g0), l14 = ld4(arow_l + g0 + 4);
    float4 l24 = ld4(arow_l + g1), l34 = ld4(arow_l + g1 + 4);
    float4 h04 = ld4(arow_h + g0), h14 = ld4(arow_h + g0 + 4);
    float4 h24 = ld4(arow_h + g1), h34 = ld4(arow_h + g1 + 4);
    const float ce[16] = {c04.x,c04.y,c04.z,c04.w, c14.x,c14.y,c14.z,c14.w,
                          c24.x,c24.y,c24.z,c24.w, c34.x,c34.y,c34.z,c34.w};
    const float le[16] = {l04.x,l04.y,l04.z,l04.w, l14.x,l14.y,l14.z,l14.w,
                          l24.x,l24.y,l24.z,l24.w, l34.x,l34.y,l34.z,l34.w};
    const float he[16] = {h04.x,h04.y,h04.z,h04.w, h14.x,h14.y,h14.z,h14.w,
                          h24.x,h24.y,h24.z,h24.w, h34.x,h34.y,h34.z,h34.w};

    // ---- B fragment: 8+8 contiguous floats (same K order as A) -> 4x b128 loads ----
    const int kB0 = kb + hsel * 8;
    float4 w04 = ld4(wrow + kB0),     w14 = ld4(wrow + kB0 + 4);
    float4 w24 = ld4(wrow + kB0 + 16), w34 = ld4(wrow + kB0 + 20);
    const float we[16] = {w04.x,w04.y,w04.z,w04.w, w14.x,w14.y,w14.z,w14.w,
                          w24.x,w24.y,w24.z,w24.w, w34.x,w34.y,w34.z,w34.w};

    v16h a0, a1, a2, b0, b2;
#pragma unroll
    for (int e = 0; e < 16; ++e) {
      a0[e] = (_Float16)ce[e];
      a1[e] = (_Float16)(0.5f * (le[e] + he[e]));
      a2[e] = (_Float16)(0.5f * (he[e] - le[e]));
      const float wv = we[e] * wscale;
      b0[e] = (_Float16)wv;
      b2[e] = (_Float16)fabsf(wv);
    }
    acc0 = __builtin_amdgcn_wmma_f32_16x16x32_f16(false, a0, false, b0,
                                                  (short)0, acc0, false, false);
    acc1 = __builtin_amdgcn_wmma_f32_16x16x32_f16(false, a1, false, b0,
                                                  (short)0, acc1, false, false);
    acc2 = __builtin_amdgcn_wmma_f32_16x16x32_f16(false, a2, false, b2,
                                                  (short)0, acc2, false, false);
  }

  // C/D layout: VGPR i holds M = i + 8*hsel, N = lane&15
  if (n < N) {
    const float bv = bias ? bias[n] : 0.f;
#pragma unroll
    for (int i = 0; i < 8; ++i) {
      const int m = i + 8 * hsel;
      float oc  = acc0[i] + bv;
      float mid = acc1[i] + bv;
      float rad = acc2[i];
      float ol = mid - rad, oh = mid + rad;
      if (do_relu) { oc = fmaxf(oc, 0.f); ol = fmaxf(ol, 0.f); oh = fmaxf(oh, 0.f); }
      const size_t d = (size_t)m * N + n;
      if (final_neg) {
        dout[d]                  = -oc;   // -c
        dout[(size_t)16 * N + d] = -oh;   // -u
        dout[(size_t)32 * N + d] = -ol;   // -l
      } else {
        cout[d] = oc; lout[d] = ol; hout[d] = oh;
      }
    }
  }
}

// ---------------------------------------------------------------------------
extern "C" void kernel_launch(void* const* d_in, const int* in_sizes, int n_in,
                              void* d_out, int out_size, void* d_ws, size_t ws_size,
                              hipStream_t stream) {
  (void)in_sizes; (void)n_in; (void)out_size; (void)ws_size;

  const float* x     = (const float*)d_in[0];
  const float* lower = (const float*)d_in[1];
  const float* upper = (const float*)d_in[2];
  const float* w1    = (const float*)d_in[3];
  const float* w2    = (const float*)d_in[4];
  const float* w3    = (const float*)d_in[5];
  const float* w4    = (const float*)d_in[6];
  const float* w5    = (const float*)d_in[7];
  const float* fw1   = (const float*)d_in[8];
  const float* fw2   = (const float*)d_in[9];
  const float* fw3   = (const float*)d_in[10];
  const float* fb3   = (const float*)d_in[11];
  float* out = (float*)d_out;

  // ping-pong triples in workspace; CAP = largest activation (16*96*15*15)
  const size_t CAP = 345600;
  float* ws = (float*)d_ws;
  float* cA = ws;            float* lA = ws + CAP;     float* hA = ws + 2 * CAP;
  float* cB = ws + 3 * CAP;  float* lB = ws + 4 * CAP; float* hB = ws + 5 * CAP;

  const dim3 blk(256);

  // conv1: [16,3,32,32] -> [16,96,15,15], k=7 s=2 pad=2
  normdist_conv_kernel<<<dim3(16 * 15 * 15, 96 / 8), blk, 0, stream>>>(
      x, lower, upper, w1, cA, lA, hA, 16, 3, 32, 32, 96, 7, 2, 2, 15, 15);

  // pool1: 15 -> 7
  {
    const int tot = 16 * 96 * 7 * 7;
    maxpool3x3s2_kernel<<<(tot + 255) / 256, blk, 0, stream>>>(
        cA, lA, hA, cB, lB, hB, 16 * 96, 15, 15, 7, 7);
  }

  // conv2: [16,96,7,7] -> [16,256,7,7], k=5 s=1 pad=2
  normdist_conv_kernel<<<dim3(16 * 7 * 7, 256 / 8), blk, 0, stream>>>(
      cB, lB, hB, w2, cA, lA, hA, 16, 96, 7, 7, 256, 5, 1, 2, 7, 7);

  // pool2: 7 -> 3
  {
    const int tot = 16 * 256 * 3 * 3;
    maxpool3x3s2_kernel<<<(tot + 255) / 256, blk, 0, stream>>>(
        cA, lA, hA, cB, lB, hB, 16 * 256, 7, 7, 3, 3);
  }

  // conv3: [16,256,3,3] -> [16,384,3,3], k=3 s=1 pad=1
  normdist_conv_kernel<<<dim3(16 * 3 * 3, 384 / 8), blk, 0, stream>>>(
      cB, lB, hB, w3, cA, lA, hA, 16, 256, 3, 3, 384, 3, 1, 1, 3, 3);

  // conv4: [16,384,3,3] -> [16,384,3,3]
  normdist_conv_kernel<<<dim3(16 * 3 * 3, 384 / 8), blk, 0, stream>>>(
      cA, lA, hA, w4, cB, lB, hB, 16, 384, 3, 3, 384, 3, 1, 1, 3, 3);

  // conv5: [16,384,3,3] -> [16,256,3,3]  (flatten -> [16, 2304])
  normdist_conv_kernel<<<dim3(16 * 3 * 3, 256 / 8), blk, 0, stream>>>(
      cB, lB, hB, w5, cA, lA, hA, 16, 384, 3, 3, 256, 3, 1, 1, 3, 3);

  // fc1: K=2304 -> N=1024, relu
  bound_linear_wmma_kernel<<<dim3((1024 / 16 + 7) / 8), blk, 0, stream>>>(
      cA, lA, hA, fw1, nullptr, cB, lB, hB, nullptr, 2304, 1024, 1, 0);

  // fc2: K=1024 -> N=512, relu
  bound_linear_wmma_kernel<<<dim3((512 / 16 + 7) / 8), blk, 0, stream>>>(
      cB, lB, hB, fw2, nullptr, cA, lA, hA, nullptr, 1024, 512, 1, 0);

  // fc3: K=512 -> N=10, +bias, final (-c, -u, -l) into d_out
  bound_linear_wmma_kernel<<<dim3(1), blk, 0, stream>>>(
      cA, lA, hA, fw3, fb3, nullptr, nullptr, nullptr, out, 512, 10, 0, 1);
}
